// GroupEmbedding_7060926234901
// MI455X (gfx1250) — matrix operand order repaired and checked by reading, not compiled, not run on hardware
//
#include <hip/hip_runtime.h>
#include <hip/hip_bf16.h>

// Problem sizes (fixed by the reference)
static constexpr int B = 2048;
static constexpr int S = 32768;
static constexpr int G = 4096;
static constexpr int E = 1024;

typedef __attribute__((ext_vector_type(16))) __bf16 v16bf;
typedef __attribute__((ext_vector_type(8)))  __bf16 v8bf;
typedef __attribute__((ext_vector_type(8)))  float  v8f;

typedef __attribute__((address_space(3))) char lds_char;

// ---------------------------------------------------------------------------
// Kernel 1: per-column metadata.
//   For permuted position s: column c = perm[s] belongs to segment g =
//   segment_ids[s]; agg mode a = agg_type[g]. Encode:
//     col2g[c] = g | (a==1 ? 0x80000000 : 0)   (OR-mode flag in sign bit)
//     colw[c]  = (a==2) ? w_elem[s] : 1.0f
// ---------------------------------------------------------------------------
__global__ __launch_bounds__(256) void k_colmeta(const int* __restrict__ perm,
                                                 const int* __restrict__ seg,
                                                 const int* __restrict__ agg,
                                                 const float* __restrict__ w_elem,
                                                 int* __restrict__ col2g,
                                                 float* __restrict__ colw) {
    int s = blockIdx.x * 256 + threadIdx.x;
    if (s >= S) return;
    int c = perm[s];
    int g = seg[s];
    int a = agg[g];
    col2g[c] = g | (a == 1 ? 0x80000000 : 0);
    colw[c]  = (a == 2) ? w_elem[s] : 1.0f;
}

// ---------------------------------------------------------------------------
// Kernel 2: W (E x G, f32) -> bf16
// ---------------------------------------------------------------------------
__global__ __launch_bounds__(256) void k_wconv(const float* __restrict__ W,
                                               __bf16* __restrict__ Wb) {
    int i = (blockIdx.x * 256 + threadIdx.x) * 4;
    if (i >= E * G) return;
    float4 v = *reinterpret_cast<const float4*>(W + i);
    Wb[i + 0] = (__bf16)v.x;
    Wb[i + 1] = (__bf16)v.y;
    Wb[i + 2] = (__bf16)v.z;
    Wb[i + 3] = (__bf16)v.w;
}

// ---------------------------------------------------------------------------
// Kernel 3: segmented aggregation. One block per row b.
//   Coalesced float4 read of row b of x (128KB), scatter into a 16KB LDS
//   f32 accumulator via ds_add_f32 atomics, then fused finalize:
//   OR-threshold (a==1), ReLU, cvt -> bf16 row of Yb (B x G).
// ---------------------------------------------------------------------------
__global__ __launch_bounds__(256) void k_segagg(const float* __restrict__ x,
                                                const int* __restrict__ col2g,
                                                const float* __restrict__ colw,
                                                const int* __restrict__ agg,
                                                __bf16* __restrict__ Yb) {
    __shared__ float acc[G];
    const int b = blockIdx.x;
    for (int i = threadIdx.x; i < G; i += 256) acc[i] = 0.0f;
    __syncthreads();

    const float4* xr = reinterpret_cast<const float4*>(x + (size_t)b * S);
    const int4*   m4 = reinterpret_cast<const int4*>(col2g);
    const float4* w4 = reinterpret_cast<const float4*>(colw);

    for (int j = threadIdx.x; j < S / 4; j += 256) {
        float4 v = xr[j];
        int4   m = m4[j];
        float4 w = w4[j];
        float c0 = (m.x < 0) ? ((v.x != 0.0f) ? 1.0f : 0.0f) : v.x * w.x;
        atomicAdd(&acc[m.x & 0xFFFF], c0);
        float c1 = (m.y < 0) ? ((v.y != 0.0f) ? 1.0f : 0.0f) : v.y * w.y;
        atomicAdd(&acc[m.y & 0xFFFF], c1);
        float c2 = (m.z < 0) ? ((v.z != 0.0f) ? 1.0f : 0.0f) : v.z * w.z;
        atomicAdd(&acc[m.z & 0xFFFF], c2);
        float c3 = (m.w < 0) ? ((v.w != 0.0f) ? 1.0f : 0.0f) : v.w * w.w;
        atomicAdd(&acc[m.w & 0xFFFF], c3);
    }
    __syncthreads();

    for (int g = threadIdx.x; g < G; g += 256) {
        float y = acc[g];
        if (agg[g] == 1) y = (y > 0.0f) ? 1.0f : 0.0f;
        y = fmaxf(y, 0.0f);   // relu
        Yb[(size_t)b * G + g] = (__bf16)y;
    }
}

// ---------------------------------------------------------------------------
// Kernel 4: GEMM  out[b,e] = sum_g Yb[b,g] * Wb[e,g]
//   M=2048, N=1024, K=4096. 128x64 block tile (grid 16x16 = 256 blocks),
//   8 waves of 32x32 (2x2 WMMA tiles), K-step 64 (two 16x16x32 slices),
//   v_wmma_f32_16x16x32_bf16.
//   Staging: double-buffered GLOBAL_LOAD_ASYNC_TO_LDS_B128 (ASYNCcnt),
//   6 async 16B chunks per thread per k-step; copies for step i+1 overlap
//   WMMA compute on step i. The LDS destination is passed to the asm as an
//   address-space(3) pointer so the buffer escapes and the compiler keeps
//   the ds_load_b128 fragment reads.
//   LDS rows padded to 144B (128B data + 16B) -> 16B-aligned b128 accesses.
// ---------------------------------------------------------------------------
static constexpr int ROWB  = 144;                    // padded LDS row stride (bytes)
static constexpr int BBASE = 128 * ROWB;             // B region offset in a buffer
static constexpr int BUFSZ = (128 + 64) * ROWB;      // A(18432)+B(9216)=27648
static constexpr int NIT   = G / 64;                 // 64 k-steps

__global__ __launch_bounds__(256) void k_gemm(const __bf16* __restrict__ A,   // Yb: B x G
                                              const __bf16* __restrict__ Wm,  // Wb: E x G
                                              float* __restrict__ out) {      // B x E
    __shared__ __attribute__((aligned(16))) char smem[2 * BUFSZ];   // 54 KB

    const int m0 = blockIdx.x * 128;
    const int n0 = blockIdx.y * 64;

    const int tid  = threadIdx.x;
    const int wave = tid >> 5;          // 0..7
    const int lane = tid & 31;
    const int wm   = wave >> 1;         // 0..3 : 32-row strip of M
    const int wn   = wave & 1;          // 0..1 : 32-col strip of N
    const int lm   = lane & 15;
    const int kh   = lane >> 4;         // 0/1

    lds_char* lbase = (lds_char*)smem;  // AS(3) base: escapes smem for the asm

    // --- per-thread async-copy assignments: 6 x 16B chunks per k-step ---
    // t=0..3 -> A tile (128 rows x 128B), t=4,5 -> B tile (64 rows x 128B)
    unsigned           ldsoff[6];
    unsigned long long ga[6];
#pragma unroll
    for (int t = 0; t < 6; ++t) {
        int q = tid + t * 256;
        if (q < 1024) {                      // A chunk
            int row = q >> 3, sg = q & 7;
            ldsoff[t] = (unsigned)(row * ROWB + sg * 16);
            ga[t] = (unsigned long long)(A + (size_t)(m0 + row) * G + sg * 8);
        } else {                             // B chunk
            int qq = q - 1024;
            int row = qq >> 3, sg = qq & 7;
            ldsoff[t] = (unsigned)(BBASE + row * ROWB + sg * 16);
            ga[t] = (unsigned long long)(Wm + (size_t)(n0 + row) * G + sg * 8);
        }
    }

    v8f acc[2][2];
#pragma unroll
    for (int mt = 0; mt < 2; ++mt)
#pragma unroll
        for (int nt = 0; nt < 2; ++nt) acc[mt][nt] = (v8f)(0.0f);

    union Frag { v16bf v; v8bf h[2]; };

    // prologue: async-fill buffer 0 with k-slice 0
#pragma unroll
    for (int t = 0; t < 6; ++t) {
        asm volatile("global_load_async_to_lds_b128 %0, %1, off"
                     :: "v"(lbase + ldsoff[t]), "v"(ga[t]) : "memory");
        ga[t] += 128;   // advance 64 bf16 in K
    }

    for (int it = 0; it < NIT; ++it) {
        const int cur = it & 1;
        const char* buf = smem + cur * BUFSZ;

        if (it + 1 < NIT) {
            // issue next k-slice into the other buffer (overlaps compute)
            const unsigned obase = (unsigned)((1 - cur) * BUFSZ);
#pragma unroll
            for (int t = 0; t < 6; ++t) {
                asm volatile("global_load_async_to_lds_b128 %0, %1, off"
                             :: "v"(lbase + (ldsoff[t] + obase)), "v"(ga[t]) : "memory");
                ga[t] += 128;
            }
            // retire the 6 older copies (targeting `cur`); 6 newest may remain
            asm volatile("s_wait_asynccnt 0x6" ::: "memory");
        } else {
            asm volatile("s_wait_asynccnt 0x0" ::: "memory");
        }
        __syncthreads();    // all waves' `cur` data visible in LDS

        // --- fragments + WMMA for two k-slices of 32 ---
#pragma unroll
        for (int kq = 0; kq < 2; ++kq) {
            // A fragments (16x32 bf16, ISA wave32 layout)
            Frag af[2];
#pragma unroll
            for (int mt = 0; mt < 2; ++mt) {
                const char* base = buf + (wm * 32 + mt * 16 + lm) * ROWB + kq * 64;
                af[mt].h[0] = *reinterpret_cast<const v8bf*>(base + kh * 16);
                af[mt].h[1] = *reinterpret_cast<const v8bf*>(base + 32 + kh * 16);
            }
            // B fragments (32x16 bf16: lane=n, kh selects K half)
            Frag bfr[2];
#pragma unroll
            for (int nt = 0; nt < 2; ++nt) {
                const char* base = buf + BBASE + (wn * 32 + nt * 16 + lm) * ROWB
                                 + kq * 64 + kh * 16;
                bfr[nt].h[0] = *reinterpret_cast<const v8bf*>(base);
                bfr[nt].h[1] = *reinterpret_cast<const v8bf*>(base + 32);
            }

#pragma unroll
            for (int mt = 0; mt < 2; ++mt)
#pragma unroll
                for (int nt = 0; nt < 2; ++nt)
                    acc[mt][nt] = __builtin_amdgcn_wmma_f32_16x16x32_bf16(
                        false, af[mt].v, false, bfr[nt].v,
                        (short)0, acc[mt][nt], false, false);
        }

        __syncthreads();    // everyone done reading `cur` before it is refilled
    }

    // --- epilogue: C/D layout -> lanes 0-15 hold M=r, lanes 16-31 M=r+8 ---
#pragma unroll
    for (int mt = 0; mt < 2; ++mt) {
#pragma unroll
        for (int nt = 0; nt < 2; ++nt) {
            int gcol  = n0 + wn * 32 + nt * 16 + lm;
            int rbase = m0 + wm * 32 + mt * 16 + kh * 8;
            v8f cc = acc[mt][nt];
#pragma unroll
            for (int r = 0; r < 8; ++r)
                out[(size_t)(rbase + r) * E + gcol] = cc[r];
        }
    }
}

// ---------------------------------------------------------------------------
// Launcher
// ---------------------------------------------------------------------------
extern "C" void kernel_launch(void* const* d_in, const int* in_sizes, int n_in,
                              void* d_out, int out_size, void* d_ws, size_t ws_size,
                              hipStream_t stream) {
    const float* x      = (const float*)d_in[0];
    const float* W      = (const float*)d_in[1];
    const float* w_elem = (const float*)d_in[2];
    const int*   perm   = (const int*)d_in[3];
    const int*   seg    = (const int*)d_in[4];
    const int*   agg    = (const int*)d_in[5];
    float* out = (float*)d_out;

    char* ws = (char*)d_ws;
    __bf16* Wb    = (__bf16*)(ws);                              // 8 MB
    __bf16* Yb    = (__bf16*)(ws + ((size_t)8 << 20));          // 16 MB
    int*    col2g = (int*)  (ws + ((size_t)24 << 20));          // 128 KB
    float*  colw  = (float*)(ws + ((size_t)24 << 20) + S * 4);  // 128 KB

    // 1) column metadata
    k_colmeta<<<(S + 255) / 256, 256, 0, stream>>>(perm, seg, agg, w_elem,
                                                   col2g, colw);
    // 2) W -> bf16
    k_wconv<<<(E * G / 4 + 255) / 256, 256, 0, stream>>>(W, Wb);
    // 3) segmented aggregation + relu -> Yb (bf16)
    k_segagg<<<B, 256, 0, stream>>>(x, col2g, colw, agg, Yb);
    // 4) relu(Y) @ W^T via bf16 WMMA, async-LDS double-buffered
    dim3 grid(B / 128, E / 64);
    k_gemm<<<grid, 256, 0, stream>>>(Yb, Wb, out);
}
